// CustomMultiHeadAttention_18468359373059
// MI455X (gfx1250) — matrix operand order repaired and checked by reading, not compiled, not run on hardware
//
#include <hip/hip_runtime.h>

// ---------------------------------------------------------------------------
// CDNA5 (gfx1250) fused multi-head attention:
//   QKV projections (+bias +RoPE) -> causal flash attention -> output proj.
// Matmuls via v_wmma_f32_16x16x32_f16 (wave32); K tiles double-buffered with
// global_load_async_to_lds_b128 (ASYNCcnt path) overlapping compute.
// ---------------------------------------------------------------------------

typedef __attribute__((ext_vector_type(16))) _Float16 v16h;
typedef __attribute__((ext_vector_type(8)))  _Float16 v8h;
typedef __attribute__((ext_vector_type(8)))  float    v8f;

#define DM    1024      // d_model
#define SEQL  1024      // sequence length
#define NH    16        // heads
#define DH    64        // head dim
#define NB    4         // batch
#define MTOT  (NB*SEQL) // 4096 rows

__device__ __forceinline__ v16h cat16(v8h lo, v8h hi) {
  v16h r;
#pragma unroll
  for (int i = 0; i < 8; ++i) { r[i] = lo[i]; r[i + 8] = hi[i]; }
  return r;
}

__device__ __forceinline__ v8f wmma16(v16h a, v16h b, v8f c) {
  // D = A(16x32 f16) * B(32x16 f16) + C(16x16 f32)
  return __builtin_amdgcn_wmma_f32_16x16x32_f16(
      /*neg_a=*/false, a, /*neg_b=*/false, b,
      /*c_mod=*/(short)0, c, /*reuse_a=*/false, /*reuse_b=*/false);
}

// ---------------------------------------------------------------------------
// GEMM: out = A[4096x1024] * W[1024x1024] + bias, f16 in, f32 acc, f16 out.
// MODE 0: store row-major [M, N]                       (output projection)
// MODE 1: store [B, H, S, Dh] with RoPE applied        (Q, K)
// MODE 2: store [B, H, S, Dh] without RoPE             (V)
// Block: 256 threads (8 waves); tile 128x64; each wave does 16 rows x 64 cols
// so every staged W tile is reused by 8 waves.
// ---------------------------------------------------------------------------
template <int MODE>
__global__ __launch_bounds__(256)
void gemm_wmma(const _Float16* __restrict__ A, const _Float16* __restrict__ W,
               const _Float16* __restrict__ bias, _Float16* __restrict__ out) {
  // W tile staged transposed: Bt[n (0..63)][k (0..31)], row stride 40 (80B, 16B-aligned)
  __shared__ __attribute__((aligned(16))) _Float16 Bt[64 * 40];

  const int tid  = threadIdx.x;
  const int wv   = tid >> 5;
  const int lane = tid & 31;
  const int rlo  = lane & 15;
  const int hi   = lane >> 4;
  const int m0   = blockIdx.x * 128;
  const int n0   = blockIdx.y * 64;
  const int mrow = m0 + wv * 16;

  float bcol[4];
#pragma unroll
  for (int t = 0; t < 4; ++t) bcol[t] = (float)bias[n0 + t * 16 + rlo];

  v8f acc[4];
#pragma unroll
  for (int t = 0; t < 4; ++t) acc[t] = (v8f){0, 0, 0, 0, 0, 0, 0, 0};

  const int kk  = tid >> 3;        // 0..31 (K within tile)
  const int nn0 = (tid & 7) * 8;   // N within tile, 8 cols per thread

  for (int kt = 0; kt < DM; kt += 32) {
    __syncthreads();
    { // stage W[kt..kt+31][n0..n0+63] transposed into LDS, coalesced 16B loads
      const _Float16* src = W + (size_t)(kt + kk) * DM + n0 + nn0;
      v8h w0 = *(const v8h*)src;
#pragma unroll
      for (int i = 0; i < 8; ++i) Bt[(nn0 + i) * 40 + kk] = w0[i];
      if (kt + 32 < DM) __builtin_prefetch(src + 32 * DM, 0, 0);
    }
    __syncthreads();

    // A fragment: lanes 0-15 hold K {kt..kt+15}; lanes 16-31 +16
    const _Float16* ap = A + (size_t)(mrow + rlo) * DM + kt + hi * 16;
    v16h af = cat16(*(const v8h*)ap, *(const v8h*)(ap + 8));

#pragma unroll
    for (int t = 0; t < 4; ++t) {
      // B fragment: lane holds column n'=rlo of subtile t, K kbase..kbase+15
      const _Float16* bp = &Bt[(t * 16 + rlo) * 40 + hi * 16];
      v16h bf = cat16(*(const v8h*)bp, *(const v8h*)(bp + 8));
      acc[t] = wmma16(af, bf, acc[t]);
    }
  }

  if (MODE == 0) {
#pragma unroll
    for (int r = 0; r < 8; ++r) {
      const int m = mrow + r + 8 * hi;
#pragma unroll
      for (int t = 0; t < 4; ++t)
        out[(size_t)m * DM + n0 + t * 16 + rlo] = (_Float16)(acc[t][r] + bcol[t]);
    }
  } else {
    const int h = blockIdx.y;  // 64-wide tiles align with heads
#pragma unroll
    for (int r = 0; r < 8; ++r) {
      const int m    = mrow + r + 8 * hi;
      const int b    = m >> 10;
      const int srow = m & (SEQL - 1);
      _Float16* ob = out + (((size_t)(b * NH + h)) * SEQL + srow) * DH;
      if (MODE == 2) {
#pragma unroll
        for (int t = 0; t < 4; ++t)
          ob[t * 16 + rlo] = (_Float16)(acc[t][r] + bcol[t]);
      } else {
        // RoPE: pair (d, d+32) -> accumulator subtiles (t, t+2), same lane
#pragma unroll
        for (int t = 0; t < 2; ++t) {
          const int j = t * 16 + rlo;  // 0..31
          const float x1 = acc[t][r]     + bcol[t];
          const float x2 = acc[t + 2][r] + bcol[t + 2];
          // inv_freq = exp(-ln(10000) * j / 32)
          const float ang = (float)srow *
                            __expf((float)j * (-9.210340371976184f / 32.0f));
          float sn, cs;
          __sincosf(ang, &sn, &cs);
          ob[j]      = (_Float16)(cs * x1 - sn * x2);
          ob[j + 32] = (_Float16)(sn * x1 + cs * x2);
        }
      }
    }
  }
}

// ---------------------------------------------------------------------------
// Causal flash attention. q/k/v layout [B,H,S,Dh]; ctx out [B,S,H*Dh].
// Grid: (S/128, B*H). Block 256 threads = 8 waves; wave owns 16 query rows;
// each staged 64-key K/V tile is reused by all 8 waves (128 queries).
// K tiles double-buffered via global_load_async_to_lds_b128: the DMA for key
// block ib+1 overlaps the V staging and the full compute of block ib.
// ---------------------------------------------------------------------------
__global__ __launch_bounds__(256)
void flash_attn(const _Float16* __restrict__ q, const _Float16* __restrict__ k,
                const _Float16* __restrict__ v, _Float16* __restrict__ ctx) {
  __shared__ __attribute__((aligned(16))) _Float16 Kt[2][64 * 72];   // [key][dh] x2
  __shared__ __attribute__((aligned(16))) _Float16 Vt[64 * 72];      // [dh][key]
  __shared__ __attribute__((aligned(16))) _Float16 Pl[8 * 16 * 72];  // per-wave P tiles

  const int tid  = threadIdx.x;
  const int wv   = tid >> 5;
  const int lane = tid & 31;
  const int rlo  = lane & 15;
  const int hi   = lane >> 4;
  const int qblk = blockIdx.x * 128;
  const int bh   = blockIdx.y;
  const size_t base = (size_t)bh * SEQL * DH;
  const int qrb  = qblk + wv * 16;

  // Q fragments (Dh split into 2 chunks of 32), pre-scaled by 1/sqrt(64)
  v16h qf[2];
#pragma unroll
  for (int c = 0; c < 2; ++c) {
    const _Float16* qp = q + base + (size_t)(qrb + rlo) * DH + c * 32 + hi * 16;
    v16h tq = cat16(*(const v8h*)qp, *(const v8h*)(qp + 8));
#pragma unroll
    for (int i = 0; i < 16; ++i) tq[i] = tq[i] * (_Float16)0.125f;
    qf[c] = tq;
  }

  float mrun[8], lrun[8];
  v8f oacc[4];
#pragma unroll
  for (int r = 0; r < 8; ++r) { mrun[r] = -1.0e30f; lrun[r] = 0.0f; }
#pragma unroll
  for (int t = 0; t < 4; ++t) oacc[t] = (v8f){0, 0, 0, 0, 0, 0, 0, 0};

  // per-thread staging geometry: thread copies 32B of one key row
  const int keyl = tid >> 2;               // key row handled by this thread
  const int coff = (tid & 3) * 32;         // byte offset within the row

  const int nkb = 2 * blockIdx.x + 2;  // causal: key blocks covering <= qblk+127

  // prologue: kick off async DMA of key block 0 into buffer 0
  {
    const unsigned l0 = (unsigned)(uintptr_t)(&Kt[0][0]) +
                        (unsigned)(keyl * 144 + coff);
    const unsigned long long g0 =
        (unsigned long long)(uintptr_t)(k + base + (size_t)keyl * DH) +
        (unsigned long long)coff;
    asm volatile(
        "global_load_async_to_lds_b128 %0, %1, off\n\t"
        "global_load_async_to_lds_b128 %0, %1, off offset:16"
        :: "v"(l0), "v"(g0) : "memory");
  }

  for (int ib = 0; ib < nkb; ++ib) {
    const int kb  = ib * 64;
    const int buf = ib & 1;

    __syncthreads();  // all waves done reading buf^1 (iter ib-1) and Vt

    if (ib + 1 < nkb) {  // async-prefetch next key block into the other buffer
      const unsigned ln = (unsigned)(uintptr_t)(&Kt[buf ^ 1][0]) +
                          (unsigned)(keyl * 144 + coff);
      const unsigned long long gn =
          (unsigned long long)(uintptr_t)(k + base +
                                          (size_t)(kb + 64 + keyl) * DH) +
          (unsigned long long)coff;
      asm volatile(
          "global_load_async_to_lds_b128 %0, %1, off\n\t"
          "global_load_async_to_lds_b128 %0, %1, off offset:16"
          :: "v"(ln), "v"(gn) : "memory");
    }

    { // stage V[kb..kb+63][0..63] transposed into LDS (overlaps async K DMA)
      const int dh0 = (tid & 3) * 16;
      const _Float16* src = v + base + (size_t)(kb + keyl) * DH + dh0;
#pragma unroll
      for (int c = 0; c < 2; ++c) {
        v8h d = *(const v8h*)(src + c * 8);
#pragma unroll
        for (int i = 0; i < 8; ++i) Vt[(dh0 + c * 8 + i) * 72 + keyl] = d[i];
      }
    }

    // async loads complete in order: <=2 outstanding leaves only block ib+1 in flight
    if (ib + 1 < nkb) asm volatile("s_wait_asynccnt 0x2" ::: "memory");
    else              asm volatile("s_wait_asynccnt 0x0" ::: "memory");
    __syncthreads();

    const _Float16* kbuf = &Kt[buf][0];

    // scores S(16x64) = Q(16x64) * K^T; K^T B-frags read from LDS (contiguous)
    v8f s[4];
#pragma unroll
    for (int t = 0; t < 4; ++t) {
      const _Float16* kp = &kbuf[(t * 16 + rlo) * 72 + hi * 16];
      v16h b0 = cat16(*(const v8h*)kp, *(const v8h*)(kp + 8));            // dh 0..31
      v16h b1 = cat16(*(const v8h*)(kp + 32), *(const v8h*)(kp + 40));    // dh 32..63
      v8f z = (v8f){0, 0, 0, 0, 0, 0, 0, 0};
      z = wmma16(qf[0], b0, z);
      s[t] = wmma16(qf[1], b1, z);
    }

    // causal mask
#pragma unroll
    for (int t = 0; t < 4; ++t) {
      const int keyA = kb + t * 16 + rlo;
#pragma unroll
      for (int r = 0; r < 8; ++r) {
        const int qA = qrb + r + 8 * hi;
        if (keyA > qA) s[t][r] = s[t][r] - 10000.0f;
      }
    }

    // online softmax: row max (across 16 lanes of each half), rescale, exp, sum
    float mnew[8], corr[8];
#pragma unroll
    for (int r = 0; r < 8; ++r) {
      float mx = fmaxf(fmaxf(s[0][r], s[1][r]), fmaxf(s[2][r], s[3][r]));
#pragma unroll
      for (int off = 1; off < 16; off <<= 1)
        mx = fmaxf(mx, __shfl_xor(mx, off, 32));
      mnew[r] = fmaxf(mrun[r], mx);
      corr[r] = __expf(mrun[r] - mnew[r]);
      mrun[r] = mnew[r];
    }
#pragma unroll
    for (int t = 0; t < 4; ++t)
#pragma unroll
      for (int r = 0; r < 8; ++r)
        s[t][r] = __expf(s[t][r] - mnew[r]);
#pragma unroll
    for (int r = 0; r < 8; ++r) {
      float sm = s[0][r] + s[1][r] + s[2][r] + s[3][r];
#pragma unroll
      for (int off = 1; off < 16; off <<= 1)
        sm += __shfl_xor(sm, off, 32);
      lrun[r] = lrun[r] * corr[r] + sm;
    }
#pragma unroll
    for (int t = 0; t < 4; ++t)
#pragma unroll
      for (int r = 0; r < 8; ++r)
        oacc[t][r] = oacc[t][r] * corr[r];

    // P: C-layout -> A-layout via per-wave LDS round-trip (intra-wave, in-order DS)
#pragma unroll
    for (int t = 0; t < 4; ++t)
#pragma unroll
      for (int r = 0; r < 8; ++r)
        Pl[(wv * 16 + r + 8 * hi) * 72 + t * 16 + rlo] = (_Float16)s[t][r];

    // ctx(16x64) += P(16x64) * V(64x64), keys contracted in 2 chunks of 32
#pragma unroll
    for (int cc = 0; cc < 2; ++cc) {
      const _Float16* pp = &Pl[(wv * 16 + rlo) * 72 + cc * 32 + hi * 16];
      v16h pa = cat16(*(const v8h*)pp, *(const v8h*)(pp + 8));
#pragma unroll
      for (int t = 0; t < 4; ++t) {
        const _Float16* vp = &Vt[(t * 16 + rlo) * 72 + cc * 32 + hi * 16];
        v16h vb = cat16(*(const v8h*)vp, *(const v8h*)(vp + 8));
        oacc[t] = wmma16(pa, vb, oacc[t]);
      }
    }
  }

  // normalize + store ctx in [B, S, H*Dh] for the output projection
  const int b = bh >> 4;
  const int h = bh & 15;
#pragma unroll
  for (int r = 0; r < 8; ++r) {
    const float inv = 1.0f / lrun[r];
    const int qA = qrb + r + 8 * hi;
    _Float16* ob = ctx + ((size_t)b * SEQL + qA) * DM + h * DH;
#pragma unroll
    for (int t = 0; t < 4; ++t)
      ob[t * 16 + rlo] = (_Float16)(oacc[t][r] * inv);
  }
}

// ---------------------------------------------------------------------------
extern "C" void kernel_launch(void* const* d_in, const int* in_sizes, int n_in,
                              void* d_out, int out_size, void* d_ws, size_t ws_size,
                              hipStream_t stream) {
  const _Float16* x  = (const _Float16*)d_in[0];
  const _Float16* Wq = (const _Float16*)d_in[1];
  const _Float16* bq = (const _Float16*)d_in[2];
  const _Float16* Wk = (const _Float16*)d_in[3];
  const _Float16* bk = (const _Float16*)d_in[4];
  const _Float16* Wv = (const _Float16*)d_in[5];
  const _Float16* bv = (const _Float16*)d_in[6];
  const _Float16* Wo = (const _Float16*)d_in[7];
  const _Float16* bo = (const _Float16*)d_in[8];
  _Float16* out = (_Float16*)d_out;

  const size_t NE = (size_t)NB * SEQL * DM;  // 4M f16 elements per buffer
  _Float16* qws = (_Float16*)d_ws;           // [B,H,S,Dh], rope'd
  _Float16* kws = qws + NE;                  // [B,H,S,Dh], rope'd
  _Float16* vws = kws + NE;                  // [B,H,S,Dh]
  _Float16* cws = vws + NE;                  // [B,S,D] context

  dim3 gg(MTOT / 128, DM / 64), gb(256);
  gemm_wmma<1><<<gg, gb, 0, stream>>>(x, Wq, bq, qws);
  gemm_wmma<1><<<gg, gb, 0, stream>>>(x, Wk, bk, kws);
  gemm_wmma<2><<<gg, gb, 0, stream>>>(x, Wv, bv, vws);

  flash_attn<<<dim3(SEQL / 128, NB * NH), 256, 0, stream>>>(qws, kws, vws, cws);

  gemm_wmma<0><<<gg, gb, 0, stream>>>(cws, Wo, bo, out);
}
